// LocallySelfAttention_90031104458856
// MI455X (gfx1250) — matrix operand order repaired and checked by reading, not compiled, not run on hardware
//
#include <hip/hip_runtime.h>
#include <hip/hip_bf16.h>

// ---------------------------------------------------------------------------
// MHA forward (pre-LN) for B=8, N=1024, E=768, H=12, D=64 on gfx1250 (CDNA5).
// All GEMMs via v_wmma_f32_16x16x32_f16 (wave32, fp32 accum).
// GEMM A-tiles are staged into LDS by the Tensor Data Mover
// (tensor_load_to_lds + s_wait_tensorcnt). V is stored transposed
// ([B][H][D][key]) so every WMMA operand load is a contiguous 32B/lane load.
// ---------------------------------------------------------------------------

#define EMBED 768
#define NHEAD 12
#define HDIM 64
#define SEQ 1024
#define BATCH 8
#define NTOK (BATCH * SEQ)   // 8192

typedef __attribute__((ext_vector_type(16))) _Float16 v16h;
typedef __attribute__((ext_vector_type(8)))  _Float16 v8h;
typedef __attribute__((ext_vector_type(8)))  float    v8f;
typedef unsigned int u32x4 __attribute__((ext_vector_type(4)));
typedef int          i32x8 __attribute__((ext_vector_type(8)));
typedef int          i32x4 __attribute__((ext_vector_type(4)));

__device__ __forceinline__ int lane_id() { return threadIdx.x & 31; }

// A fragment: 16x32 f16 from row-major [16 x >=32], ld in elements.
__device__ __forceinline__ v16h load_frag_a(const _Float16* __restrict__ p, int ld) {
  const int lane = lane_id();
  const int m = lane & 15;
  const int kh = (lane >> 4) << 4;  // 0 or 16
  const _Float16* r = p + m * ld + kh;
  v16h a;
#pragma unroll
  for (int i = 0; i < 8; ++i) a[i] = r[i];
#pragma unroll
  for (int i = 0; i < 8; ++i) a[8 + i] = r[8 + i];
  return a;
}

// B fragment where logical B = S^T and S is row-major [N][K]: B[k][n] = S[n][k].
__device__ __forceinline__ v16h load_frag_b_tr(const _Float16* __restrict__ p, int ld) {
  const int lane = lane_id();
  const int n = lane & 15;
  const int k0 = (lane >> 4) << 4;
  v16h b;
  __builtin_memcpy(&b, p + n * ld + k0, sizeof(v16h));
  return b;
}

__device__ __forceinline__ v8f wmma_f16(v16h a, v16h b, v8f c) {
  return __builtin_amdgcn_wmma_f32_16x16x32_f16(false, a, false, b, (short)0, c,
                                                false, false);
}

// ---------------------------------------------------------------------------
// Tensor Data Mover: DMA a 64-row x 32-f16 tile (row stride 768 f16) from
// global into LDS (packed 64x32). D# packed per CDNA5 ISA §8.3-8.5.
// Issue from a single wave; completion via s_wait_tensorcnt.
// Toolchain uses the 6-arg builtin: (g0, g1, g2, g3, extra, cpol).
// ---------------------------------------------------------------------------
__device__ __forceinline__ void tdm_load_tile_64x32(const _Float16* gsrc,
                                                    unsigned lds_byte_addr) {
  const unsigned long long ga = (unsigned long long)(size_t)gsrc;
  u32x4 g0;
  g0[0] = 1u;                                           // count=1 (valid user D#)
  g0[1] = lds_byte_addr;                                // lds_addr (bytes)
  g0[2] = (unsigned)(ga & 0xFFFFFFFFu);                 // global_addr[31:0]
  g0[3] = (unsigned)((ga >> 32) & 0x1FFFFFFu)           // global_addr[56:32]
          | (2u << 30);                                 // type=2 ("image")
  i32x8 g1;
  g1[0] = (int)(1u << 16);                              // wg_mask=0, data_size=1 (2B)
  g1[1] = (int)((768u & 0xFFFFu) << 16);                // tensor_dim0[15:0] @63:48
  g1[2] = (int)((768u >> 16) | ((8192u & 0xFFFFu) << 16)); // dim0 hi | tensor_dim1 lo
  g1[3] = (int)((8192u >> 16) | (32u << 16));           // dim1 hi | tile_dim0=32
  g1[4] = 64;                                           // tile_dim1=64, tile_dim2=0
  g1[5] = 768;                                          // tensor_dim0_stride[31:0]
  g1[6] = 0;                                            // stride0 hi | stride1 lo
  g1[7] = 0;
  const i32x4 gz4 = {0, 0, 0, 0};                       // 2D: groups 2/3 unused
  const i32x8 gz8 = {0, 0, 0, 0, 0, 0, 0, 0};
  __builtin_amdgcn_tensor_load_to_lds(g0, g1, gz4, gz4, gz8, 0);
}

// ---------------------------------------------------------------------------
// f32 -> f16 convert (weights)
// ---------------------------------------------------------------------------
__global__ __launch_bounds__(256) void cvt_f16_kernel(const float* __restrict__ src,
                                                      _Float16* __restrict__ dst,
                                                      int n) {
  int i = blockIdx.x * 256 + threadIdx.x;
  if (i < n) dst[i] = (_Float16)src[i];
}

// ---------------------------------------------------------------------------
// Fused LayerNorm -> f16. One block (256 threads) per token row of 768.
// ---------------------------------------------------------------------------
__global__ __launch_bounds__(256) void ln_f16_kernel(const float* __restrict__ x,
                                                     const float* __restrict__ gamma,
                                                     const float* __restrict__ beta,
                                                     _Float16* __restrict__ xn) {
  const int row = blockIdx.x;
  const int tid = threadIdx.x;
  const float* xr = x + (size_t)row * EMBED;
  float v[3], s = 0.f, ss = 0.f;
#pragma unroll
  for (int i = 0; i < 3; ++i) {
    v[i] = xr[tid + 256 * i];
    s += v[i];
    ss += v[i] * v[i];
  }
#pragma unroll
  for (int off = 1; off < 32; off <<= 1) {
    s += __shfl_xor(s, off, 32);
    ss += __shfl_xor(ss, off, 32);
  }
  __shared__ float red_s[8], red_ss[8];
  if ((tid & 31) == 0) { red_s[tid >> 5] = s; red_ss[tid >> 5] = ss; }
  __syncthreads();
  float ts = 0.f, tss = 0.f;
#pragma unroll
  for (int i = 0; i < 8; ++i) { ts += red_s[i]; tss += red_ss[i]; }
  const float mu = ts * (1.0f / EMBED);
  const float var = tss * (1.0f / EMBED) - mu * mu;
  const float rstd = rsqrtf(var + 1e-5f);
  _Float16* outr = xn + (size_t)row * EMBED;
#pragma unroll
  for (int i = 0; i < 3; ++i) {
    const int c = tid + 256 * i;
    outr[c] = (_Float16)((v[i] - mu) * rstd * gamma[c] + beta[c]);
  }
}

// ---------------------------------------------------------------------------
// QKV GEMM: qkv[m][f] = xn[m][:] . w_qkv[f][:]
// A tile (64x32) staged by TDM; W^T fragments streamed from global/L2.
// Scatter: q (pre-scaled by 1/tau) and k as [B][H][N][D]; v as Vt[B][H][D][key].
// ---------------------------------------------------------------------------
__global__ __launch_bounds__(256) void gemm_qkv_kernel(
    const _Float16* __restrict__ xn, const _Float16* __restrict__ wq,
    const float* __restrict__ tau_p, _Float16* __restrict__ qh,
    _Float16* __restrict__ kh, _Float16* __restrict__ vt) {
  __shared__ __align__(16) _Float16 lds_a[64 * 32];
  const int tid = threadIdx.x;
  const int wave = tid >> 5;
  const int wx = wave & 3;
  const int wy = wave >> 2;
  const int m_base = blockIdx.x * 64;
  const int n_base = blockIdx.y * 128;
  const float inv_tau = 1.0f / tau_p[0];
  const unsigned lds_base = (unsigned)(size_t)(&lds_a[0]);
  if (tid == 0) ((volatile _Float16*)lds_a)[0] = (_Float16)0.f;  // keep LDS "live"
  v8f acc[2][2] = {};
  for (int k0 = 0; k0 < EMBED; k0 += 32) {
    if (wave == 0) {
      tdm_load_tile_64x32(xn + (size_t)m_base * EMBED + k0, lds_base);
      __builtin_amdgcn_s_wait_tensorcnt(0);
    }
    __syncthreads();
    const v16h a0 = load_frag_a(&lds_a[(wy * 32) * 32], 32);
    const v16h a1 = load_frag_a(&lds_a[(wy * 32 + 16) * 32], 32);
    const v16h b0 = load_frag_b_tr(wq + (size_t)(n_base + wx * 32) * EMBED + k0, EMBED);
    const v16h b1 = load_frag_b_tr(wq + (size_t)(n_base + wx * 32 + 16) * EMBED + k0, EMBED);
    acc[0][0] = wmma_f16(a0, b0, acc[0][0]);
    acc[0][1] = wmma_f16(a0, b1, acc[0][1]);
    acc[1][0] = wmma_f16(a1, b0, acc[1][0]);
    acc[1][1] = wmma_f16(a1, b1, acc[1][1]);
    __syncthreads();
  }
  const int lane = tid & 31;
  const int ncol = lane & 15;
  const int m_off = (lane >> 4) << 3;
#pragma unroll
  for (int my = 0; my < 2; ++my)
#pragma unroll
    for (int nb = 0; nb < 2; ++nb)
#pragma unroll
      for (int e = 0; e < 8; ++e) {
        const int m = m_base + wy * 32 + my * 16 + m_off + e;
        const int f = n_base + wx * 32 + nb * 16 + ncol;
        const int which = f / EMBED;  // 0:q 1:k 2:v
        const int fe = f % EMBED;
        const int h = fe >> 6, d = fe & 63;
        const int b = m >> 10, t = m & 1023;
        float val = acc[my][nb][e];
        if (which == 0) val *= inv_tau;  // fold 1/tau into q in fp32
        const _Float16 hv = (_Float16)val;
        if (which == 2) {
          vt[((((size_t)b * NHEAD + h) * HDIM + d) << 10) + t] = hv;  // Vt[b][h][d][t]
        } else {
          const size_t idx = ((((size_t)b * NHEAD + h) << 10) + t) * HDIM + d;
          if (which == 0) qh[idx] = hv; else kh[idx] = hv;
        }
      }
}

// ---------------------------------------------------------------------------
// Flash attention: one wave owns 16 query rows of one (b,h); streams keys in
// chunks of 64 with online softmax (q pre-scaled by 1/tau).
// ---------------------------------------------------------------------------
__global__ __launch_bounds__(256) void attn_kernel(
    const _Float16* __restrict__ qh, const _Float16* __restrict__ kh,
    const _Float16* __restrict__ vt, const unsigned char* __restrict__ mask,
    _Float16* __restrict__ ctx) {
  __shared__ __align__(16) _Float16 lds_p[8][16 * 64];
  const int tid = threadIdx.x;
  const int lane = tid & 31;
  const int wave = tid >> 5;
  const int job = blockIdx.x * 8 + wave;              // 6144 jobs total
  const int qb = job & 63;                            // query 16-block
  const int h = (job >> 6) % NHEAD;
  const int b = job / (64 * NHEAD);
  const size_t bh = (((size_t)b * NHEAD + h) << 10) * HDIM;   // q/k base
  const size_t bhd = (((size_t)b * NHEAD + h) * HDIM) << 10;  // vt base
  const int q0 = qb * 16;
  const v16h aq0 = load_frag_a(qh + bh + (size_t)q0 * HDIM, HDIM);
  const v16h aq1 = load_frag_a(qh + bh + (size_t)q0 * HDIM + 32, HDIM);
  v8f acc[4] = {};
  float mrow[8], lrow[8];
#pragma unroll
  for (int e = 0; e < 8; ++e) { mrow[e] = -3.0e38f; lrow[e] = 0.f; }
  const int ncol = lane & 15;
  const int m_off = (lane >> 4) << 3;
  const unsigned char* mb = mask + ((size_t)b * SEQ + q0) * SEQ;

  for (int key0 = 0; key0 < SEQ; key0 += 64) {
    // ---- scores S[16 x 64] = Q . K^T (8 WMMAs) ----
    v8f s[4] = {};
#pragma unroll
    for (int nb = 0; nb < 4; ++nb) {
      const _Float16* kbp = kh + bh + (size_t)(key0 + nb * 16) * HDIM;
      const v16h b0 = load_frag_b_tr(kbp, HDIM);
      const v16h b1 = load_frag_b_tr(kbp + 32, HDIM);
      s[nb] = wmma_f16(aq0, b0, s[nb]);
      s[nb] = wmma_f16(aq1, b1, s[nb]);
    }
    if (key0 + 64 < SEQ) {
      __builtin_prefetch(kh + bh + (size_t)(key0 + 64 + (lane & 31)) * HDIM, 0, 1);
      __builtin_prefetch(vt + bhd + (size_t)(lane & 31) * SEQ + key0 + 64, 0, 1);
    }
    // ---- mask + online softmax ----
    float corr[8];
#pragma unroll
    for (int e = 0; e < 8; ++e) {
      const unsigned char* mp = mb + (size_t)(m_off + e) * SEQ + key0 + ncol;
      float sv[4];
#pragma unroll
      for (int c = 0; c < 4; ++c) {
        sv[c] = s[c][e];
        if (mp[c * 16]) sv[c] = -3.0e38f;
      }
      float mx = fmaxf(fmaxf(sv[0], sv[1]), fmaxf(sv[2], sv[3]));
#pragma unroll
      for (int off = 1; off < 16; off <<= 1) mx = fmaxf(mx, __shfl_xor(mx, off, 32));
      const float mn = fmaxf(mrow[e], mx);
      const float cf = __expf(mrow[e] - mn);
      float ps = 0.f;
#pragma unroll
      for (int c = 0; c < 4; ++c) {
        const float p = __expf(sv[c] - mn);
        s[c][e] = p;
        ps += p;
      }
#pragma unroll
      for (int off = 1; off < 16; off <<= 1) ps += __shfl_xor(ps, off, 32);
      lrow[e] = lrow[e] * cf + ps;
      mrow[e] = mn;
      corr[e] = cf;
    }
#pragma unroll
    for (int j = 0; j < 4; ++j)
#pragma unroll
      for (int e = 0; e < 8; ++e) acc[j][e] *= corr[e];
    // ---- P: C-layout -> A-layout via per-wave LDS slab (in-order DS) ----
    _Float16* pp = &lds_p[wave][0];
#pragma unroll
    for (int e = 0; e < 8; ++e)
#pragma unroll
      for (int c = 0; c < 4; ++c)
        pp[(m_off + e) * 64 + c * 16 + ncol] = (_Float16)s[c][e];
    const v16h ap0 = load_frag_a(pp, 64);
    const v16h ap1 = load_frag_a(pp + 32, 64);
    // ---- O += P . V  (8 WMMAs; Vt gives contiguous B fragments) ----
#pragma unroll
    for (int j = 0; j < 4; ++j) {
      const _Float16* vp = vt + bhd + ((size_t)(j * 16) << 10) + key0;
      const v16h bv0 = load_frag_b_tr(vp, SEQ);
      const v16h bv1 = load_frag_b_tr(vp + 32, SEQ);
      acc[j] = wmma_f16(ap0, bv0, acc[j]);
      acc[j] = wmma_f16(ap1, bv1, acc[j]);
    }
  }
#pragma unroll
  for (int j = 0; j < 4; ++j)
#pragma unroll
    for (int e = 0; e < 8; ++e) {
      const int row = q0 + m_off + e;
      const float o = acc[j][e] / lrow[e];
      const size_t idx = (((size_t)b * SEQ + row) * EMBED) + h * HDIM + j * 16 + ncol;
      ctx[idx] = (_Float16)o;
    }
}

// ---------------------------------------------------------------------------
// Output GEMM: out[m][f] = ctx[m][:] . w_o[f][:] + b_o[f], fp32 output.
// A tile staged by TDM, same as QKV GEMM.
// ---------------------------------------------------------------------------
__global__ __launch_bounds__(256) void gemm_o_kernel(
    const _Float16* __restrict__ ctx, const _Float16* __restrict__ wo,
    const float* __restrict__ bo, float* __restrict__ out) {
  __shared__ __align__(16) _Float16 lds_a[64 * 32];
  const int tid = threadIdx.x;
  const int wave = tid >> 5;
  const int wx = wave & 3;
  const int wy = wave >> 2;
  const int m_base = blockIdx.x * 64;
  const int n_base = blockIdx.y * 128;
  const unsigned lds_base = (unsigned)(size_t)(&lds_a[0]);
  if (tid == 0) ((volatile _Float16*)lds_a)[0] = (_Float16)0.f;  // keep LDS "live"
  v8f acc[2][2] = {};
  for (int k0 = 0; k0 < EMBED; k0 += 32) {
    if (wave == 0) {
      tdm_load_tile_64x32(ctx + (size_t)m_base * EMBED + k0, lds_base);
      __builtin_amdgcn_s_wait_tensorcnt(0);
    }
    __syncthreads();
    const v16h a0 = load_frag_a(&lds_a[(wy * 32) * 32], 32);
    const v16h a1 = load_frag_a(&lds_a[(wy * 32 + 16) * 32], 32);
    const v16h b0 = load_frag_b_tr(wo + (size_t)(n_base + wx * 32) * EMBED + k0, EMBED);
    const v16h b1 = load_frag_b_tr(wo + (size_t)(n_base + wx * 32 + 16) * EMBED + k0, EMBED);
    acc[0][0] = wmma_f16(a0, b0, acc[0][0]);
    acc[0][1] = wmma_f16(a0, b1, acc[0][1]);
    acc[1][0] = wmma_f16(a1, b0, acc[1][0]);
    acc[1][1] = wmma_f16(a1, b1, acc[1][1]);
    __syncthreads();
  }
  const int lane = tid & 31;
  const int ncol = lane & 15;
  const int m_off = (lane >> 4) << 3;
#pragma unroll
  for (int my = 0; my < 2; ++my)
#pragma unroll
    for (int nb = 0; nb < 2; ++nb)
#pragma unroll
      for (int e = 0; e < 8; ++e) {
        const int m = m_base + wy * 32 + my * 16 + m_off + e;
        const int f = n_base + wx * 32 + nb * 16 + ncol;
        out[(size_t)m * EMBED + f] = acc[my][nb][e] + bo[f];
      }
}

// ---------------------------------------------------------------------------
extern "C" void kernel_launch(void* const* d_in, const int* in_sizes, int n_in,
                              void* d_out, int out_size, void* d_ws, size_t ws_size,
                              hipStream_t stream) {
  (void)in_sizes; (void)n_in; (void)out_size; (void)ws_size;
  const float* x = (const float*)d_in[0];
  const unsigned char* mask = (const unsigned char*)d_in[1];
  const float* ln_scale = (const float*)d_in[2];
  const float* ln_bias = (const float*)d_in[3];
  const float* tau = (const float*)d_in[4];
  const float* w_qkv = (const float*)d_in[5];
  const float* w_o = (const float*)d_in[6];
  const float* b_o = (const float*)d_in[7];
  float* out = (float*)d_out;

  char* ws = (char*)d_ws;
  size_t off = 0;
  auto alloc = [&](size_t bytes) -> char* {
    char* p = ws + off;
    off += (bytes + 255) & ~(size_t)255;
    return p;
  };
  _Float16* xn_h   = (_Float16*)alloc((size_t)NTOK * EMBED * 2);
  _Float16* wqkv_h = (_Float16*)alloc((size_t)3 * EMBED * EMBED * 2);
  _Float16* wo_h   = (_Float16*)alloc((size_t)EMBED * EMBED * 2);
  _Float16* q_h    = (_Float16*)alloc((size_t)NTOK * EMBED * 2);
  _Float16* k_h    = (_Float16*)alloc((size_t)NTOK * EMBED * 2);
  _Float16* v_t    = (_Float16*)alloc((size_t)NTOK * EMBED * 2);
  _Float16* ctx_h  = (_Float16*)alloc((size_t)NTOK * EMBED * 2);

  const int nwq = 3 * EMBED * EMBED;
  const int nwo = EMBED * EMBED;
  cvt_f16_kernel<<<(nwq + 255) / 256, 256, 0, stream>>>(w_qkv, wqkv_h, nwq);
  cvt_f16_kernel<<<(nwo + 255) / 256, 256, 0, stream>>>(w_o, wo_h, nwo);
  ln_f16_kernel<<<NTOK, 256, 0, stream>>>(x, ln_scale, ln_bias, xn_h);
  gemm_qkv_kernel<<<dim3(NTOK / 64, (3 * EMBED) / 128), 256, 0, stream>>>(
      xn_h, wqkv_h, tau, q_h, k_h, v_t);
  attn_kernel<<<(BATCH * NHEAD * (SEQ / 16)) / 8, 256, 0, stream>>>(
      q_h, k_h, v_t, mask, ctx_h);
  gemm_o_kernel<<<dim3(NTOK / 64, EMBED / 128), 256, 0, stream>>>(
      ctx_h, wo_h, b_o, out);
}
